// OrionDecoupledDoubleStreamAttnProcessor2_0_60017872994705
// MI455X (gfx1250) — compile-verified, matrix operand order checked by reading, not run on hardware
//
#include <hip/hip_runtime.h>
#include <hip/hip_bf16.h>
#include <math.h>

#define SB_IMG 2304
#define SB_TXT 512
#define SB_TOT 2816
#define DIM    3072
#define NH     24
#define HDIM   128

typedef unsigned short u16;
typedef unsigned int   u32;
typedef __attribute__((ext_vector_type(16))) __bf16 v16bf;
typedef __attribute__((ext_vector_type(8)))  float  v8f;

union FragA { v16bf v; u32 u[8]; };
union FragC { v8f   v; float f[8]; };

static __device__ __forceinline__ u16 f2bf(float f){
#if defined(__gfx1250__)
  __bf16 h = (__bf16)f;              // native v_cvt (RNE)
  return __builtin_bit_cast(u16, h);
#else
  u32 x = __float_as_uint(f);
  x += 0x7FFFu + ((x >> 16) & 1u);
  return (u16)(x >> 16);
#endif
}
static __device__ __forceinline__ u32 pack2(float lo, float hi){
  return (u32)f2bf(lo) | ((u32)f2bf(hi) << 16);
}
static __device__ __forceinline__ v8f wmma_bf16(v16bf a, v16bf b, v8f c){
  // D(f32,16x16) = A(bf16,16x32) * B(bf16,32x16) + C
  return __builtin_amdgcn_wmma_f32_16x16x32_bf16(false, a, false, b, (short)0, c, false, false);
}

// CDNA5 async global->LDS copy (16B per lane, tracked by ASYNCcnt).
static __device__ __forceinline__ void async_cp16(u16* dst_lds, const u16* src_glob){
#if defined(__gfx1250__)
  u32 l = (u32)(uintptr_t)dst_lds;   // LDS aperture: low 32 bits == DS offset
  unsigned long long g = (unsigned long long)(uintptr_t)src_glob;
  asm volatile("global_load_async_to_lds_b128 %0, %1, off"
               :: "v"(l), "v"(g) : "memory");
#else
  *(uint4*)dst_lds = *(const uint4*)src_glob;
#endif
}
static __device__ __forceinline__ void async_wait0(){
#if defined(__gfx1250__)
  asm volatile("s_wait_asynccnt 0x0" ::: "memory");
#endif
}

// Load a 16x32 bf16 A/B fragment from a row-major LDS tile.
// CDNA5 16-bit layout: lanes 0-15 row=lane, K 0..7 (V0-3) & 16..23 (V4-7);
// lanes 16-31 same rows, K 8..15 & 24..31; 2 elements packed per VGPR.
static __device__ __forceinline__ void load_frag(FragA& f, const u16* base, int stride, int lane, int koff){
  const u16* r = base + (lane & 15) * stride + koff + ((lane >> 4) << 3);
  #pragma unroll
  for (int v = 0; v < 8; ++v){
    int k = ((v & 4) ? 16 : 0) + ((v & 3) * 2);
    f.u[v] = *(const u32*)(r + k);
  }
}

// ---------------------------------------------------------------------------
// GEMM: C[64x128] = A[64xK] * W[128xK]^T (+bias), K = 3072.
// EPI 0: bias -> bf16 head-major QKV buffer (V path)
// EPI 1: bias -> RMS norm -> RoPE -> bf16 head-major QKV buffer (Q/K path)
// EPI 2: bias -> fp32 row-major output (final projections)
// ---------------------------------------------------------------------------
#define AS_S 48
#define WS_S 48
#define CS_S 132

template<bool AF32, int EPI>
__global__ __launch_bounds__(256) void gemm_k(
    const void* __restrict__ Ap,
    const float* __restrict__ W,
    const float* __restrict__ bias,
    u16* __restrict__ outb, int s_offset,
    float* __restrict__ outf,
    const float* __restrict__ rmsw,
    const float* __restrict__ cosp,
    const float* __restrict__ sinp)
{
  __shared__ __align__(16) u16   As[64 * AS_S];
  __shared__ __align__(16) u16   Ws[128 * WS_S];
  __shared__ __align__(16) float Cs[64 * CS_S];
  __shared__ __align__(16) float Rs[64 * 4];

  const int tid  = threadIdx.x;
  const int lane = tid & 31;
  const int wave = tid >> 5;
  const int wm   = wave & 3;   // 16-row strip
  const int wn   = wave >> 2;  // 64-col half
  const int m0   = blockIdx.x * 64;
  const int n0   = blockIdx.y * 128;

  FragC acc[4];
  #pragma unroll
  for (int t = 0; t < 4; ++t)
    #pragma unroll
    for (int v = 0; v < 8; ++v) acc[t].f[v] = 0.0f;

  const int ar = tid >> 2, ac = (tid & 3) * 8;    // A loader map (64x32)
  const int wr = tid >> 1, wc = (tid & 1) * 16;   // W loader map (128x32)

  for (int k0 = 0; k0 < DIM; k0 += 32){
    __syncthreads();
    if constexpr (AF32){
      const float* Ag = (const float*)Ap + (size_t)(m0 + ar) * DIM + k0 + ac;
      float4 x0 = *(const float4*)(Ag);
      float4 x1 = *(const float4*)(Ag + 4);
      u32* d = (u32*)&As[ar * AS_S + ac];
      d[0] = pack2(x0.x, x0.y); d[1] = pack2(x0.z, x0.w);
      d[2] = pack2(x1.x, x1.y); d[3] = pack2(x1.z, x1.w);
    } else {
      // bf16 A tile: raw 16B/lane copy -> async global->LDS (ASYNCcnt path)
      const u16* Ag = (const u16*)Ap + (size_t)(m0 + ar) * DIM + k0 + ac;
      async_cp16(&As[ar * AS_S + ac], Ag);
    }
    {
      const float* Wg = W + (size_t)(n0 + wr) * DIM + k0 + wc;
      u32* d = (u32*)&Ws[wr * WS_S + wc];
      #pragma unroll
      for (int j = 0; j < 4; ++j){
        float4 x = *(const float4*)(Wg + 4 * j);
        d[2 * j]     = pack2(x.x, x.y);
        d[2 * j + 1] = pack2(x.z, x.w);
      }
      if (k0 + 32 < DIM) __builtin_prefetch(Wg + 32, 0, 0);
    }
    if constexpr (!AF32) async_wait0();
    __syncthreads();

    FragA a;
    load_frag(a, &As[wm * 16 * AS_S], AS_S, lane, 0);
    #pragma unroll
    for (int t = 0; t < 4; ++t){
      FragA b;
      load_frag(b, &Ws[(wn * 64 + t * 16) * WS_S], WS_S, lane, 0);
      acc[t].v = wmma_bf16(a.v, b.v, acc[t].v);
    }
  }

  __syncthreads();
  {
    // C layout: VGPR v -> row v (lanes 0-15) / row v+8 (lanes 16-31), col = lane&15
    const int rowHi = (lane >> 4) << 3;
    #pragma unroll
    for (int t = 0; t < 4; ++t)
      #pragma unroll
      for (int v = 0; v < 8; ++v)
        Cs[(wm * 16 + v + rowHi) * CS_S + wn * 64 + t * 16 + (lane & 15)] = acc[t].f[v];
  }
  __syncthreads();

  // Epilogue: 4 threads per row, 32 columns each.
  const int tr = tid >> 2;
  const int tc = (tid & 3) * 32;
  float x[32];
  #pragma unroll
  for (int j = 0; j < 32; ++j)
    x[j] = Cs[tr * CS_S + tc + j] + bias[n0 + tc + j];

  if constexpr (EPI == 2){
    float* o = outf + (size_t)(m0 + tr) * DIM + n0 + tc;
    #pragma unroll
    for (int j = 0; j < 32; ++j) o[j] = x[j];
  } else if constexpr (EPI == 1){
    float s = 0.0f;
    #pragma unroll
    for (int j = 0; j < 32; ++j) s += x[j] * x[j];
    Rs[tr * 4 + (tid & 3)] = s;
    __syncthreads();
    float tot = Rs[tr * 4 + 0] + Rs[tr * 4 + 1] + Rs[tr * 4 + 2] + Rs[tr * 4 + 3];
    float inv = rsqrtf(tot * (1.0f / 128.0f) + 1e-6f);
    const int h = blockIdx.y;                 // BN == HDIM: one head per block col
    const float* cr = cosp + (size_t)(m0 + tr) * (HDIM / 2);
    const float* sr = sinp + (size_t)(m0 + tr) * (HDIM / 2);
    u16* o = outb + ((size_t)h * SB_TOT + (size_t)(s_offset + m0 + tr)) * HDIM + tc;
    #pragma unroll
    for (int j = 0; j < 32; j += 2){
      float a  = x[j]     * inv * rmsw[tc + j];
      float b  = x[j + 1] * inv * rmsw[tc + j + 1];
      float c  = cr[(tc + j) >> 1];
      float sn = sr[(tc + j) >> 1];
      o[j]     = f2bf(a * c - b * sn);
      o[j + 1] = f2bf(a * sn + b * c);
    }
  } else {
    const int h = blockIdx.y;
    u16* o = outb + ((size_t)h * SB_TOT + (size_t)(s_offset + m0 + tr)) * HDIM + tc;
    #pragma unroll
    for (int j = 0; j < 32; ++j) o[j] = f2bf(x[j]);
  }
}

// ---------------------------------------------------------------------------
// Flash attention: grid (S_TOT/128, H); each wave owns 16 query rows.
// Q/K/V are [H][S_TOT][HDIM] bf16; output is [S_TOT][DIM] bf16.
// Q and K tiles staged via async global->LDS; V transposed through VGPRs.
// ---------------------------------------------------------------------------
#define QS_S 144
#define KS_S 144
#define VT_S 68
#define PS_S 68

__global__ __launch_bounds__(256) void attn_k(
    const u16* __restrict__ Q, const u16* __restrict__ K,
    const u16* __restrict__ V, u16* __restrict__ O)
{
  __shared__ __align__(16) u16 Qs[128 * QS_S];
  __shared__ __align__(16) u16 Ks[64 * KS_S];
  __shared__ __align__(16) u16 Vt[128 * VT_S];   // transposed: [d][k]
  __shared__ __align__(16) u16 Ps[8 * 16 * PS_S];

  const int tid  = threadIdx.x;
  const int lane = tid & 31;
  const int wave = tid >> 5;
  const int qb   = blockIdx.x;
  const int h    = blockIdx.y;
  const int mr   = wave * 16;

  { // Q tile: 128 x 128 bf16, async copy
    const u16* Qg = Q + ((size_t)h * SB_TOT + (size_t)qb * 128) * HDIM;
    for (int i = tid; i < 128 * 16; i += 256){
      int r = i >> 4, c = (i & 15) * 8;
      async_cp16(&Qs[r * QS_S + c], Qg + r * HDIM + c);
    }
  }

  float mrow[8], lrow[8];
  FragC o[8];
  #pragma unroll
  for (int v = 0; v < 8; ++v){ mrow[v] = -INFINITY; lrow[v] = 0.0f; }
  #pragma unroll
  for (int t = 0; t < 8; ++t)
    #pragma unroll
    for (int v = 0; v < 8; ++v) o[t].f[v] = 0.0f;

  const float scale = 0.08838834764831845f;   // 1/sqrt(128)

  for (int kt = 0; kt < SB_TOT / 64; ++kt){
    __syncthreads();
    const u16* Kg = K + ((size_t)h * SB_TOT + (size_t)kt * 64) * HDIM;
    for (int i = tid; i < 64 * 16; i += 256){
      int r = i >> 4, c = (i & 15) * 8;
      async_cp16(&Ks[r * KS_S + c], Kg + r * HDIM + c);
    }
    const u16* Vg = V + ((size_t)h * SB_TOT + (size_t)kt * 64) * HDIM;
    for (int i = tid; i < 64 * 64; i += 256){
      int r = i >> 6, c2 = (i & 63) * 2;
      u32 val = *(const u32*)(Vg + r * HDIM + c2);
      Vt[c2 * VT_S + r]       = (u16)(val & 0xFFFFu);
      Vt[(c2 + 1) * VT_S + r] = (u16)(val >> 16);
    }
    async_wait0();          // covers Q (first iter) + K tile async copies
    __syncthreads();

    // S = Q * K^T  (16 x 64 per wave)
    FragC sf[4];
    #pragma unroll
    for (int t = 0; t < 4; ++t)
      #pragma unroll
      for (int v = 0; v < 8; ++v) sf[t].f[v] = 0.0f;
    #pragma unroll
    for (int ks = 0; ks < 4; ++ks){
      FragA a;
      load_frag(a, &Qs[mr * QS_S], QS_S, lane, ks * 32);
      #pragma unroll
      for (int t = 0; t < 4; ++t){
        FragA b;
        load_frag(b, &Ks[t * 16 * KS_S], KS_S, lane, ks * 32);
        sf[t].v = wmma_bf16(a.v, b.v, sf[t].v);
      }
    }

    // Online softmax: each row lives in one 16-lane half -> xor 1,2,4,8 reduces it.
    float corr[8];
    #pragma unroll
    for (int v = 0; v < 8; ++v){
      float mx = sf[0].f[v];
      #pragma unroll
      for (int t = 1; t < 4; ++t) mx = fmaxf(mx, sf[t].f[v]);
      mx *= scale;
      #pragma unroll
      for (int d = 1; d <= 8; d <<= 1) mx = fmaxf(mx, __shfl_xor(mx, d, 32));
      float mnew = fmaxf(mrow[v], mx);
      corr[v] = __expf(mrow[v] - mnew);
      mrow[v] = mnew;
      float rs = 0.0f;
      #pragma unroll
      for (int t = 0; t < 4; ++t){
        float p = __expf(sf[t].f[v] * scale - mnew);
        sf[t].f[v] = p;
        rs += p;
      }
      #pragma unroll
      for (int d = 1; d <= 8; d <<= 1) rs += __shfl_xor(rs, d, 32);
      lrow[v] = lrow[v] * corr[v] + rs;
    }

    #pragma unroll
    for (int t = 0; t < 8; ++t)
      #pragma unroll
      for (int v = 0; v < 8; ++v) o[t].f[v] *= corr[v];

    { // restage P (C-layout) into A-fragment layout via LDS
      const int rowHi = (lane >> 4) << 3;
      u16* pw = &Ps[wave * 16 * PS_S];
      #pragma unroll
      for (int t = 0; t < 4; ++t)
        #pragma unroll
        for (int v = 0; v < 8; ++v)
          pw[(v + rowHi) * PS_S + t * 16 + (lane & 15)] = f2bf(sf[t].f[v]);
    }
    __syncthreads();

    // O += P * V   (16 x 128 per wave)
    #pragma unroll
    for (int ks = 0; ks < 2; ++ks){
      FragA pa;
      load_frag(pa, &Ps[wave * 16 * PS_S], PS_S, lane, ks * 32);
      #pragma unroll
      for (int t = 0; t < 8; ++t){
        FragA vb;
        load_frag(vb, &Vt[t * 16 * VT_S], VT_S, lane, ks * 32);
        o[t].v = wmma_bf16(pa.v, vb.v, o[t].v);
      }
    }
  }

  { // normalize + store to [S_TOT][DIM]
    const int rowHi = (lane >> 4) << 3;
    float inv[8];
    #pragma unroll
    for (int v = 0; v < 8; ++v) inv[v] = 1.0f / lrow[v];
    #pragma unroll
    for (int t = 0; t < 8; ++t)
      #pragma unroll
      for (int v = 0; v < 8; ++v){
        int row = qb * 128 + mr + v + rowHi;
        int col = h * HDIM + t * 16 + (lane & 15);
        O[(size_t)row * DIM + col] = f2bf(o[t].f[v] * inv[v]);
      }
  }
}

extern "C" void kernel_launch(void* const* d_in, const int* in_sizes, int n_in,
                              void* d_out, int out_size, void* d_ws, size_t ws_size,
                              hipStream_t stream) {
  (void)in_sizes; (void)n_in; (void)out_size; (void)ws_size;
  const float* hidden  = (const float*)d_in[0];
  const float* enc     = (const float*)d_in[1];
  const float* wq  = (const float*)d_in[2];  const float* bq  = (const float*)d_in[3];
  const float* wk  = (const float*)d_in[4];  const float* bk  = (const float*)d_in[5];
  const float* wv  = (const float*)d_in[6];  const float* bv  = (const float*)d_in[7];
  const float* waq = (const float*)d_in[8];  const float* baq = (const float*)d_in[9];
  const float* wak = (const float*)d_in[10]; const float* bak = (const float*)d_in[11];
  const float* wav = (const float*)d_in[12]; const float* bav = (const float*)d_in[13];
  const float* nq  = (const float*)d_in[14]; const float* nk  = (const float*)d_in[15];
  const float* naq = (const float*)d_in[16]; const float* nak = (const float*)d_in[17];
  const float* w_out     = (const float*)d_in[18]; const float* b_out     = (const float*)d_in[19];
  const float* w_add_out = (const float*)d_in[20]; const float* b_add_out = (const float*)d_in[21];
  const float* img_cos = (const float*)d_in[22]; const float* img_sin = (const float*)d_in[23];
  const float* txt_cos = (const float*)d_in[24]; const float* txt_sin = (const float*)d_in[25];

  const size_t qkvElems = (size_t)NH * SB_TOT * HDIM;
  u16* Qb = (u16*)d_ws;
  u16* Kb = Qb + qkvElems;
  u16* Vb = Kb + qkvElems;
  u16* Ab = Vb + qkvElems;          // attention output, [S_TOT][DIM] bf16
  float* out = (float*)d_out;

  dim3 blk(256);
  // img stream QKV (rows land at s = 512..2815)
  gemm_k<true, 1><<<dim3(SB_IMG / 64, NH), blk, 0, stream>>>(hidden, wq, bq, Qb, SB_TXT, nullptr, nq, img_cos, img_sin);
  gemm_k<true, 1><<<dim3(SB_IMG / 64, NH), blk, 0, stream>>>(hidden, wk, bk, Kb, SB_TXT, nullptr, nk, img_cos, img_sin);
  gemm_k<true, 0><<<dim3(SB_IMG / 64, NH), blk, 0, stream>>>(hidden, wv, bv, Vb, SB_TXT, nullptr, nullptr, nullptr, nullptr);
  // txt stream QKV (rows land at s = 0..511)
  gemm_k<true, 1><<<dim3(SB_TXT / 64, NH), blk, 0, stream>>>(enc, waq, baq, Qb, 0, nullptr, naq, txt_cos, txt_sin);
  gemm_k<true, 1><<<dim3(SB_TXT / 64, NH), blk, 0, stream>>>(enc, wak, bak, Kb, 0, nullptr, nak, txt_cos, txt_sin);
  gemm_k<true, 0><<<dim3(SB_TXT / 64, NH), blk, 0, stream>>>(enc, wav, bav, Vb, 0, nullptr, nullptr, nullptr, nullptr);
  // joint attention over S_TOT = 2816
  attn_k<<<dim3(SB_TOT / 128, NH), blk, 0, stream>>>(Qb, Kb, Vb, Ab);
  // output projections: img rows 512.., txt rows 0..511
  gemm_k<false, 2><<<dim3(SB_IMG / 64, DIM / 128), blk, 0, stream>>>(
      Ab + (size_t)SB_TXT * DIM, w_out, b_out, nullptr, 0, out, nullptr, nullptr, nullptr);
  gemm_k<false, 2><<<dim3(SB_TXT / 64, DIM / 128), blk, 0, stream>>>(
      Ab, w_add_out, b_add_out, nullptr, 0, out + (size_t)SB_IMG * DIM, nullptr, nullptr, nullptr);
}
